// DCMSABlock_3221225472164
// MI455X (gfx1250) — matrix operand order repaired and checked by reading, not compiled, not run on hardware
//
#include <hip/hip_runtime.h>
#include <hip/hip_bf16.h>
#include <math.h>

// ---------------- problem constants (from reference) ----------------
#define BB     2
#define TT     2048
#define DD     512
#define HH     8
#define KK     16          // keys per query (dilated window)
#define DEPTH  3
#define ROWS   (BB*TT)     // 4096
#define HD     (DD/HH)     // 64

typedef __attribute__((ext_vector_type(16))) _Float16 v16h;
typedef __attribute__((ext_vector_type(8)))  _Float16 v8h;
typedef __attribute__((ext_vector_type(8)))  float    v8f;

// gfx1250 async memory->LDS path (ASYNCcnt-tracked), guarded so the compile
// can never regress: fall back to global_load + ds_store if builtins absent.
#if __has_builtin(__builtin_amdgcn_global_load_async_to_lds_b128) && \
    __has_builtin(__builtin_amdgcn_s_wait_asynccnt)
#define USE_ASYNC 1
typedef __attribute__((ext_vector_type(4))) int i4v;
typedef __attribute__((address_space(1))) i4v* gp_t;   // global int4*
typedef __attribute__((address_space(3))) i4v* lp_t;   // LDS int4*
#else
#define USE_ASYNC 0
#endif

__device__ __forceinline__ void tile_cp16(const _Float16* g, _Float16* l) {
#if USE_ASYNC
    // generic LDS pointer: low 32 bits are the LDS offset (aperture scheme)
    __builtin_amdgcn_global_load_async_to_lds_b128(
        (gp_t)(uintptr_t)g, (lp_t)(unsigned int)(uintptr_t)l, 0, 0);
#else
    *(uint4*)l = *(const uint4*)g;
#endif
}

__device__ __forceinline__ void tile_wait() {
#if USE_ASYNC
    __builtin_amdgcn_s_wait_asynccnt(0);
#endif
}

// ---------------- weight convert + transpose:  in (K x N) f32 -> out (N x K) f16
__global__ __launch_bounds__(256)
void wconv_kernel(const float* __restrict__ in, _Float16* __restrict__ out,
                  int Kd, int Nd) {
    size_t idx = (size_t)blockIdx.x * 256 + threadIdx.x;
    if (idx >= (size_t)Kd * Nd) return;
    int k = (int)(idx / Nd);
    int n = (int)(idx % Nd);
    out[(size_t)n * Kd + k] = (_Float16)in[idx];
}

// ---------------- LayerNorm: x f32 (rows x 512) -> h f16, one wave per row
__global__ __launch_bounds__(256)
void ln_kernel(const float* __restrict__ x, const float* __restrict__ s,
               const float* __restrict__ b, _Float16* __restrict__ h) {
    const int lane = threadIdx.x & 31;
    const int row  = blockIdx.x * 8 + (threadIdx.x >> 5);
    const float* xr = x + (size_t)row * DD;

    float v[16];
    float sum = 0.f;
#pragma unroll
    for (int j = 0; j < 16; ++j) { v[j] = xr[lane + j * 32]; sum += v[j]; }
#pragma unroll
    for (int m = 16; m; m >>= 1) sum += __shfl_xor(sum, m, 32);
    const float mean = sum * (1.f / DD);

    float var = 0.f;
#pragma unroll
    for (int j = 0; j < 16; ++j) { float d = v[j] - mean; var += d * d; }
#pragma unroll
    for (int m = 16; m; m >>= 1) var += __shfl_xor(var, m, 32);
    const float inv = rsqrtf(var * (1.f / DD) + 1e-5f);

    _Float16* hr = h + (size_t)row * DD;
#pragma unroll
    for (int j = 0; j < 16; ++j) {
        const int c = lane + j * 32;
        hr[c] = (_Float16)((v[j] - mean) * inv * s[c] + b[c]);
    }
}

// ---------------- dilated-window attention: one wave per (b,h,query)
// qkv: (ROWS x 1536) f32, row (b*T+t), col (three*H + h)*64 + e
// out: (ROWS x 512)  f16, col h*64 + e
__global__ __launch_bounds__(256)
void attn_kernel(const float* __restrict__ qkv, _Float16* __restrict__ o, int dil) {
    const int lane = threadIdx.x & 31;
    const int gw   = (blockIdx.x << 3) + (threadIdx.x >> 5);   // [0, B*H*T)
    const int b    = gw >> 14;            // H*T = 16384
    const int rem  = gw & 16383;
    const int h    = rem >> 11;           // T = 2048
    const int i    = rem & 2047;

    const size_t rowQ = (size_t)(b * TT + i) * (3 * DD);
    const int    hc   = h * HD;
    const float2 q    = *(const float2*)(qkv + rowQ + hc + lane * 2);

    float sc[KK];
#pragma unroll
    for (int j = 0; j < KK; ++j) {
        const int p = i - j * dil;
        float d = -1e30f;
        if (p >= 0) {
            const float2 kk = *(const float2*)(qkv + (size_t)(b * TT + p) * (3 * DD)
                                               + DD + hc + lane * 2);
            d = q.x * kk.x + q.y * kk.y;
#pragma unroll
            for (int m = 16; m; m >>= 1) d += __shfl_xor(d, m, 32);
            d *= 0.125f;                  // hd^-0.5
        }
        sc[j] = d;
    }
    float mx = sc[0];
#pragma unroll
    for (int j = 1; j < KK; ++j) mx = fmaxf(mx, sc[j]);
    float den = 0.f;
#pragma unroll
    for (int j = 0; j < KK; ++j) { sc[j] = __expf(sc[j] - mx); den += sc[j]; }
    const float invden = 1.f / den;

    float ax = 0.f, ay = 0.f;
#pragma unroll
    for (int j = 0; j < KK; ++j) {
        const int p = i - j * dil;
        if (p >= 0) {
            const float2 vv = *(const float2*)(qkv + (size_t)(b * TT + p) * (3 * DD)
                                               + 2 * DD + hc + lane * 2);
            ax += sc[j] * vv.x;  ay += sc[j] * vv.y;
        }
    }
    _Float16* orow = o + (size_t)(b * TT + i) * DD + hc + lane * 2;
    orow[0] = (_Float16)(ax * invden);
    orow[1] = (_Float16)(ay * invden);
}

// ---------------- WMMA GEMM:  C(MxN) = A(MxK,f16,row) * Bt(NxK,f16,row)^T
// block tile 128x128, 8 waves (4x2), each wave 32x64 -> 2x4 fragments,
// 8 x v_wmma_f32_16x16x32_f16 per 32-k-step, double-buffered LDS (32 KB),
// async global->LDS tile staging when available.
#define EPI_NONE      0   // store f32 (ldc = N)
#define EPI_BIAS_RES  1   // f32 out = acc + bias[n] + res[row,n]
#define EPI_BIAS_GELU 2   // f16 out = gelu(acc + bias[n])

template <int EPI>
__global__ __launch_bounds__(256)
void gemm_wmma(const _Float16* __restrict__ A, const _Float16* __restrict__ Bt,
               float* __restrict__ Cf, _Float16* __restrict__ Ch,
               const float* __restrict__ bias, const float* __restrict__ res,
               int M, int N, int K) {
    __shared__ __align__(16) _Float16 sA[2][128 * 32];
    __shared__ __align__(16) _Float16 sB[2][128 * 32];

    const int tid  = threadIdx.x;
    const int lane = tid & 31;
    const int wid  = tid >> 5;
    const int wm   = wid >> 1;            // 0..3  (rows of 32)
    const int wn   = wid & 1;             // 0..1  (cols of 64)
    const int bm   = blockIdx.y * 128;
    const int bn   = blockIdx.x * 128;

    const int halfrow = lane & 15;
    const int kgrp    = lane >> 4;        // 0/1

    const int lr = tid >> 2;              // 0..63
    const int lq = (tid & 3) * 8;         // 0,8,16,24 halves

    v8f acc[2][4] = {};

    auto load_tiles = [&](int kt, int bf) {
        tile_cp16(&A [(size_t)(bm + lr     ) * K + kt + lq], &sA[bf][(lr     ) * 32 + lq]);
        tile_cp16(&A [(size_t)(bm + lr + 64) * K + kt + lq], &sA[bf][(lr + 64) * 32 + lq]);
        tile_cp16(&Bt[(size_t)(bn + lr     ) * K + kt + lq], &sB[bf][(lr     ) * 32 + lq]);
        tile_cp16(&Bt[(size_t)(bn + lr + 64) * K + kt + lq], &sB[bf][(lr + 64) * 32 + lq]);
    };

    load_tiles(0, 0);
    tile_wait();
    __syncthreads();

    for (int kt = 0; kt < K; kt += 32) {
        const int  cur  = (kt >> 5) & 1;
        const bool more = (kt + 32) < K;
        if (more) load_tiles(kt + 32, cur ^ 1);

        v16h a[2], b[4];
#pragma unroll
        for (int i = 0; i < 2; ++i) {
            // A frag 16x32: lane<16 -> K {0..15}; lane>=16 -> K {16..31}
            const _Float16* p = &sA[cur][(wm * 32 + i * 16 + halfrow) * 32 + kgrp * 16];
            v8h lo = *(const v8h*)p;
            v8h hi = *(const v8h*)(p + 8);
            a[i] = __builtin_shufflevector(lo, hi, 0,1,2,3,4,5,6,7,8,9,10,11,12,13,14,15);
        }
#pragma unroll
        for (int j = 0; j < 4; ++j) {
            // B frag 32x16: lane holds column (lane&15), 16 contiguous K at kgrp*16
            const _Float16* p = &sB[cur][(wn * 64 + j * 16 + halfrow) * 32 + kgrp * 16];
            v8h lo = *(const v8h*)p;
            v8h hi = *(const v8h*)(p + 8);
            b[j] = __builtin_shufflevector(lo, hi, 0,1,2,3,4,5,6,7,8,9,10,11,12,13,14,15);
        }
#pragma unroll
        for (int i = 0; i < 2; ++i)
#pragma unroll
            for (int j = 0; j < 4; ++j)
                acc[i][j] = __builtin_amdgcn_wmma_f32_16x16x32_f16(
                    false, a[i], false, b[j], (short)0, acc[i][j], false, false);

        if (more) tile_wait();            // my wave's async loads landed
        __syncthreads();                  // everyone done reading cur / loads visible
    }

    // C layout: VGPR r, lane<16 -> (M=r, N=lane); lane>=16 -> (M=r+8, N=lane-16)
    const int rbase = bm + wm * 32 + (kgrp << 3);
    const int cbase = bn + wn * 64 + halfrow;
#pragma unroll
    for (int i = 0; i < 2; ++i) {
#pragma unroll
        for (int j = 0; j < 4; ++j) {
            const int col = cbase + j * 16;
            const float bv = (EPI != EPI_NONE) ? bias[col] : 0.f;
#pragma unroll
            for (int r = 0; r < 8; ++r) {
                const int row = rbase + i * 16 + r;
                float v = acc[i][j][r];
                if (EPI == EPI_NONE) {
                    Cf[(size_t)row * N + col] = v;
                } else if (EPI == EPI_BIAS_RES) {
                    Cf[(size_t)row * N + col] = v + bv + res[(size_t)row * N + col];
                } else {
                    v += bv;
                    v = 0.5f * v * (1.f + erff(v * 0.70710678118654752f));
                    Ch[(size_t)row * N + col] = (_Float16)v;
                }
            }
        }
    }
}

// ---------------- host-side orchestration ----------------
static inline size_t align256(size_t x) { return (x + 255) & ~(size_t)255; }

extern "C" void kernel_launch(void* const* d_in, const int* in_sizes, int n_in,
                              void* d_out, int out_size, void* d_ws, size_t ws_size,
                              hipStream_t stream) {
    (void)in_sizes; (void)n_in; (void)out_size; (void)ws_size;
    const float* x      = (const float*)d_in[0];
    const float* ln1_s  = (const float*)d_in[1];
    const float* ln1_b  = (const float*)d_in[2];
    const float* qkv_w  = (const float*)d_in[3];   // (DEPTH, 512, 1536)
    const float* proj_w = (const float*)d_in[4];   // (DEPTH, 512, 512)
    const float* proj_b = (const float*)d_in[5];
    const float* ln2_s  = (const float*)d_in[6];
    const float* ln2_b  = (const float*)d_in[7];
    const float* w1     = (const float*)d_in[8];   // (DEPTH, 512, 2048)
    const float* b1     = (const float*)d_in[9];
    const float* w2     = (const float*)d_in[10];  // (DEPTH, 2048, 512)
    const float* b2     = (const float*)d_in[11];
    float* xout = (float*)d_out;                   // running residual x (4096 x 512)

    // workspace carve-out
    char* w = (char*)d_ws;
    _Float16* qkvT = (_Float16*)w;  w += align256((size_t)DEPTH * 1536 * 512 * 2);
    _Float16* projT= (_Float16*)w;  w += align256((size_t)DEPTH * 512  * 512 * 2);
    _Float16* w1T  = (_Float16*)w;  w += align256((size_t)DEPTH * 2048 * 512 * 2);
    _Float16* w2T  = (_Float16*)w;  w += align256((size_t)DEPTH * 512  * 2048 * 2);
    _Float16* h16  = (_Float16*)w;  w += align256((size_t)ROWS * DD * 2);
    _Float16* a16  = (_Float16*)w;  w += align256((size_t)ROWS * DD * 2);
    _Float16* m16  = (_Float16*)w;  w += align256((size_t)ROWS * 2048 * 2);
    float*    qkvf = (float*)w;     w += align256((size_t)ROWS * 1536 * 4);

    // convert + transpose all weights to f16 (N x K)
    for (int d = 0; d < DEPTH; ++d) {
        wconv_kernel<<<(512 * 1536 + 255) / 256, 256, 0, stream>>>(
            qkv_w + (size_t)d * 512 * 1536, qkvT + (size_t)d * 1536 * 512, 512, 1536);
        wconv_kernel<<<(512 * 512 + 255) / 256, 256, 0, stream>>>(
            proj_w + (size_t)d * 512 * 512, projT + (size_t)d * 512 * 512, 512, 512);
        wconv_kernel<<<(512 * 2048 + 255) / 256, 256, 0, stream>>>(
            w1 + (size_t)d * 512 * 2048, w1T + (size_t)d * 2048 * 512, 512, 2048);
        wconv_kernel<<<(2048 * 512 + 255) / 256, 256, 0, stream>>>(
            w2 + (size_t)d * 2048 * 512, w2T + (size_t)d * 512 * 2048, 2048, 512);
    }

    // x -> out (residual accumulator lives in d_out)
    (void)hipMemcpyAsync(xout, x, (size_t)ROWS * DD * sizeof(float),
                         hipMemcpyDeviceToDevice, stream);

    for (int d = 0; d < DEPTH; ++d) {
        // h = LN1(x)
        ln_kernel<<<ROWS / 8, 256, 0, stream>>>(xout, ln1_s + d * DD, ln1_b + d * DD, h16);
        // qkv = h @ qkv_w
        gemm_wmma<EPI_NONE><<<dim3(1536 / 128, ROWS / 128), 256, 0, stream>>>(
            h16, qkvT + (size_t)d * 1536 * 512, qkvf, nullptr, nullptr, nullptr,
            ROWS, 1536, DD);
        // dilated attention
        attn_kernel<<<(BB * HH * TT) / 8, 256, 0, stream>>>(qkvf, a16, 1 << d);
        // x += attn_out @ proj_w + proj_b
        gemm_wmma<EPI_BIAS_RES><<<dim3(512 / 128, ROWS / 128), 256, 0, stream>>>(
            a16, projT + (size_t)d * 512 * 512, xout, nullptr, proj_b + d * DD, xout,
            ROWS, DD, DD);
        // h = LN2(x)
        ln_kernel<<<ROWS / 8, 256, 0, stream>>>(xout, ln2_s + d * DD, ln2_b + d * DD, h16);
        // m = gelu(h @ w1 + b1)  (f16)
        gemm_wmma<EPI_BIAS_GELU><<<dim3(2048 / 128, ROWS / 128), 256, 0, stream>>>(
            h16, w1T + (size_t)d * 2048 * 512, nullptr, m16, b1 + (size_t)d * 2048, nullptr,
            ROWS, 2048, DD);
        // x += m @ w2 + b2
        gemm_wmma<EPI_BIAS_RES><<<dim3(512 / 128, ROWS / 128), 256, 0, stream>>>(
            m16, w2T + (size_t)d * 512 * 2048, xout, nullptr, b2 + d * DD, xout,
            ROWS, DD, 2048);
    }
}
